// GNNModel_60894046322824
// MI455X (gfx1250) — matrix operand order repaired and checked by reading, not compiled, not run on hardware
//
#include <hip/hip_runtime.h>
#include <hip/hip_bf16.h>

// ---------------------------------------------------------------------------
// Model constants (from the reference)
// ---------------------------------------------------------------------------
#define BB   65536
#define NN   5
#define EE   20
#define NF_  7
#define EF_  2
#define DD   16
#define EDD  4
#define LL   4
#define EPSF 1e-5f
#define SLOPEF 0.01f

// MLP head layer shapes (K, Kpad, N)
//  L0: 80 ->512   (pad K to 96)
//  L1: 512->1024
//  L2: 1024->1024
//  L3: 1024->1024
//  L4: 1024->128
//  L5: 128->16 collapsed with mean into a 128-dot
#define STR   1032              // LDS row stride in bf16 elems (1024 + 8 pad)
#define MTILE 64                // batch rows per head workgroup
#define BUFB  (MTILE * STR * 2) // one LDS activation buffer, bytes
#define HEAD_LDS (2 * BUFB)

typedef __attribute__((ext_vector_type(16))) __bf16 v16bf;
typedef __attribute__((ext_vector_type(8)))  float  v8f;

struct __attribute__((aligned(16))) U4 { unsigned int x, y, z, w; };
struct __attribute__((aligned(16))) U8 { U4 a, b; };

__device__ __forceinline__ unsigned short f2bf(float f) {
    return __builtin_bit_cast(unsigned short, (__bf16)f);   // hw RNE convert
}
__device__ __forceinline__ float bf2f(unsigned short h) {
    unsigned int u = ((unsigned int)h) << 16;
    return __builtin_bit_cast(float, u);
}
// Build a 16xbf16 fragment from two 16-byte chunks.
__device__ __forceinline__ v16bf ld_frag(const void* p0, const void* p1) {
    U8 t;
    t.a = *(const U4*)p0;
    t.b = *(const U4*)p1;
    return __builtin_bit_cast(v16bf, t);
}
// slope < 1  =>  leaky_relu(x) == max(x, slope*x): one v_max instead of cmp+sel
__device__ __forceinline__ float lrelu(float x) {
    return fmaxf(x, SLOPEF * x);
}

// ---------------------------------------------------------------------------
// Weight prep: fp32 [K,N] -> bf16 tiled B-fragment layout in ws.
// Tile (kt,nt) holds K in [kt*32,kt*32+32), N in [nt*16,nt*16+16).
// dst = (kt*(N/16)+nt)*512 + nl*32 + kl  (kl contiguous => per-lane b128 loads)
// Also folds linear bias + BN into scale/shift: y = acc*scale[n] + shift[n].
// ---------------------------------------------------------------------------
__global__ __launch_bounds__(256) void prep_layer(
    const float* __restrict__ W, const float* __restrict__ bias,
    const float* __restrict__ g, const float* __restrict__ b,
    const float* __restrict__ m, const float* __restrict__ v,
    unsigned short* __restrict__ Wt, float* __restrict__ scale,
    float* __restrict__ shift, int K, int Kpad, int N) {
    int j = blockIdx.x * 256 + threadIdx.x;
    int total = Kpad * N;
    if (j < total) {
        int k = j / N, n = j % N;
        float w = (k < K) ? W[(long)k * N + n] : 0.f;
        int kt = k >> 5, kl = k & 31, nt = n >> 4, nl = n & 15;
        Wt[(((long)kt * (N >> 4) + nt) << 9) + nl * 32 + kl] = f2bf(w);
    } else if (j < total + N) {
        int n = j - total;
        float s = g[n] * rsqrtf(v[n] + EPSF);
        scale[n] = s;
        shift[n] = (bias[n] - m[n]) * s + b[n];
    }
}

// lin5 (128->16) + mean over 16  ==>  out = dot(act, w5r) + c5
__global__ __launch_bounds__(256) void prep_final(
    const float* __restrict__ W5, const float* __restrict__ b5,
    float* __restrict__ w5r) {
    int t = threadIdx.x;
    if (t < 128) {
        float s = 0.f;
        for (int n = 0; n < 16; ++n) s += W5[t * 16 + n];
        w5r[t] = s * (1.f / 16.f);
    } else if (t == 128) {
        float s = 0.f;
        for (int n = 0; n < 16; ++n) s += b5[n];
        w5r[128] = s * (1.f / 16.f);
    }
}

// ---------------------------------------------------------------------------
// GNN part: embeddings + 4 conv layers, fp32 VALU, LDS-staged per 32 rows.
// Deterministic aggregation (fixed edge order). Output: bf16 [B,80] in ws.
// ---------------------------------------------------------------------------
struct GArgs {
    const float* nf; const float* ef; const int* ei;
    const float* new_; const float* neb;
    const float* eew;  const float* eeb;
    const float* conv[LL][16];   // msg1.w,b  msg2.w,b  upd1.w,b  bn1.g,b,m,v  upd2.w,b  bn2.g,b,m,v
    unsigned short* xout;
};

__global__ __launch_bounds__(256) void gnn_kernel(GArgs A) {
    __shared__ float xs[32][NN * DD];        // 10 KB
    __shared__ float eas[32][EE][EDD];       // 10 KB
    __shared__ float ms[32][EE][DD];         // 40 KB
    __shared__ float wb[1704];               // per-layer conv weights
    __shared__ int ij[EE], ii[EE];
    int tid = threadIdx.x;
    long b0 = (long)blockIdx.x * 32;

    if (tid < EE) { ij[tid] = A.ei[tid]; ii[tid] = A.ei[EE + tid]; }

    // node embedding: (b,n) -> 16
    if (tid < 32 * NN) {
        int b = tid / NN, n = tid % NN;
        const float* f = A.nf + (b0 + b) * (NN * NF_) + n * NF_;
        for (int o = 0; o < DD; ++o) {
            float acc = A.neb[o];
            for (int i = 0; i < NF_; ++i) acc += f[i] * A.new_[i * DD + o];
            xs[b][n * DD + o] = acc;
        }
    }
    // edge embedding: (b,e) -> 4
    for (int p = tid; p < 32 * EE; p += 256) {
        int b = p / EE, e = p % EE;
        const float* f = A.ef + (b0 + b) * (EE * EF_) + e * EF_;
        for (int c = 0; c < EDD; ++c)
            eas[b][e][c] = A.eeb[c] + f[0] * A.eew[c] + f[1] * A.eew[EDD + c];
    }
    __syncthreads();

    const int segsz[16] = {576, 16, 256, 16, 512, 16, 5, 5, 5, 5, 256, 16, 5, 5, 5, 5};

    for (int l = 0; l < LL; ++l) {
        {   // stage this layer's weights in LDS
            int off = 0;
            for (int s = 0; s < 16; ++s) {
                for (int i = tid; i < segsz[s]; i += 256) wb[off + i] = A.conv[l][s][i];
                off += segsz[s];
            }
        }
        __syncthreads();
        const float* W1 = wb;        const float* b1 = wb + 576;
        const float* W2 = wb + 592;  const float* b2 = wb + 848;
        // message phase: (b,e)
        for (int p = tid; p < 32 * EE; p += 256) {
            int b = p / EE, e = p % EE;
            const float* hi = &xs[b][ii[e] * DD];
            const float* hj = &xs[b][ij[e] * DD];
            const float* ev = eas[b][e];
            float m1[DD];
            for (int o = 0; o < DD; ++o) {
                float acc = b1[o];
                for (int i = 0; i < DD;  ++i) acc += hi[i] * W1[i * DD + o];
                for (int i = 0; i < DD;  ++i) acc += hj[i] * W1[(DD + i) * DD + o];
                for (int i = 0; i < EDD; ++i) acc += ev[i] * W1[(2 * DD + i) * DD + o];
                m1[o] = lrelu(acc);
            }
            for (int o = 0; o < DD; ++o) {
                float acc = b2[o];
                for (int i = 0; i < DD; ++i) acc += m1[i] * W2[i * DD + o];
                ms[b][e][o] = lrelu(acc);
            }
        }
        __syncthreads();
        // aggregate + update phase: (b,n)
        const float* Wu1 = wb + 864;  const float* bu1 = wb + 1376;
        const float* bn1 = wb + 1392;                 // g,b,m,v x5
        const float* Wu2 = wb + 1412; const float* bu2 = wb + 1668;
        const float* bn2 = wb + 1684;
        if (tid < 32 * NN) {
            int b = tid / NN, n = tid % NN;
            float agg[DD];
            for (int o = 0; o < DD; ++o) agg[o] = 0.f;
            for (int e = 0; e < EE; ++e)
                if (ii[e] == n)
                    for (int o = 0; o < DD; ++o) agg[o] += ms[b][e][o];
            float* xp = &xs[b][n * DD];
            float s1  = bn1[n]      * rsqrtf(bn1[3 * NN + n] + EPSF);
            float sh1 = bn1[NN + n] - bn1[2 * NN + n] * s1;
            float s2  = bn2[n]      * rsqrtf(bn2[3 * NN + n] + EPSF);
            float sh2 = bn2[NN + n] - bn2[2 * NN + n] * s2;
            float u1[DD];
            for (int o = 0; o < DD; ++o) {
                float acc = bu1[o];
                for (int i = 0; i < DD; ++i) acc += xp[i]  * Wu1[i * DD + o];
                for (int i = 0; i < DD; ++i) acc += agg[i] * Wu1[(DD + i) * DD + o];
                u1[o] = lrelu(acc * s1 + sh1);
            }
            float u2[DD];
            for (int o = 0; o < DD; ++o) {
                float acc = bu2[o];
                for (int i = 0; i < DD; ++i) acc += u1[i] * Wu2[i * DD + o];
                u2[o] = lrelu(acc * s2 + sh2);
            }
            for (int o = 0; o < DD; ++o) xp[o] += u2[o];   // residual
        }
        __syncthreads();
    }
    // emit bf16 rows [B, 80]
    for (int i = tid; i < 32 * NN * DD; i += 256) {
        int b = i / (NN * DD), j = i % (NN * DD);
        A.xout[(b0 + b) * (NN * DD) + j] = f2bf(xs[b][j]);
    }
}

// ---------------------------------------------------------------------------
// MLP head: fused 5 WMMA layers + final 128-dot, activations in LDS ping-pong.
// One workgroup = 8 wave32 = 64 batch rows. Each wave computes a 32x32 output
// block (2x2 WMMA tiles): 4 independent accumulators per k-step -> latency
// hiding, no D->A/B hazards, and 1/2 the LDS+L2 traffic per v_wmma.
// ---------------------------------------------------------------------------
template <int KP, int NO>
__device__ __forceinline__ void mlp_layer(
    char* smem, int inB, int outB,
    const unsigned short* __restrict__ Wt,
    const float* __restrict__ sc, const float* __restrict__ sh,
    int lane, int wave) {
    const int lg = lane >> 4;        // lane group (0: lanes 0-15, 1: 16-31)
    const int l15 = lane & 15;
    const int NT = NO >> 4;          // number of 16-wide N tiles
    // Supertiles: (mtp in 0..1) x (ntp in 0..NT/2-1); wave-uniform loop so
    // EXEC stays all-ones for WMMA.
#pragma unroll 1
    for (int t = wave; t < NT; t += 8) {
        int mtp = t & 1, ntp = t >> 1;
        v8f a00 = {}, a01 = {}, a10 = {}, a11 = {};
        const char* arow0 = smem + inB + (mtp * 32 + l15) * (STR * 2)
                            + (lg * 8) * 2;
        const unsigned short* wb0 = Wt + ((long)(2 * ntp) << 9) + l15 * 32 + lg * 8;
#pragma unroll 1
        for (int k0 = 0; k0 < KP / 32; ++k0) {
            // A fragments (16-bit A layout): elems 0-7 = K k0*32+lg*8.., 8-15 = +16
            const char* ap0 = arow0 + k0 * 64;
            const char* ap1 = ap0 + 16 * (STR * 2);
            v16bf a0 = ld_frag(ap0, ap0 + 32);
            v16bf a1 = ld_frag(ap1, ap1 + 32);
            // B fragments: elems 0-7 = K lg*8.., 8-15 = +16, at fixed column
            const unsigned short* bp0 = wb0 + (long)k0 * NT * 512;
            const unsigned short* bp1 = bp0 + 512;
            v16bf b0 = ld_frag(bp0, bp0 + 16);
            v16bf b1 = ld_frag(bp1, bp1 + 16);
            a00 = __builtin_amdgcn_wmma_f32_16x16x32_bf16(false, a0, false, b0, (short)0, a00, false, false);
            a01 = __builtin_amdgcn_wmma_f32_16x16x32_bf16(false, a0, false, b1, (short)0, a01, false, false);
            a10 = __builtin_amdgcn_wmma_f32_16x16x32_bf16(false, a1, false, b0, (short)0, a10, false, false);
            a11 = __builtin_amdgcn_wmma_f32_16x16x32_bf16(false, a1, false, b1, (short)0, a11, false, false);
        }
        int col0 = ntp * 32 + l15, col1 = col0 + 16;
        float s0 = sc[col0], o0 = sh[col0];
        float s1 = sc[col1], o1 = sh[col1];
        int r0 = mtp * 32 + lg * 8;       // C layout: lanes16-31 hold M=r+8
#pragma unroll
        for (int r = 0; r < 8; ++r) {
            char* d0 = smem + outB + ((r0 + r) * STR) * 2;
            char* d1 = d0 + 16 * (STR * 2);
            *(unsigned short*)(d0 + col0 * 2) = f2bf(lrelu(a00[r] * s0 + o0));
            *(unsigned short*)(d0 + col1 * 2) = f2bf(lrelu(a01[r] * s1 + o1));
            *(unsigned short*)(d1 + col0 * 2) = f2bf(lrelu(a10[r] * s0 + o0));
            *(unsigned short*)(d1 + col1 * 2) = f2bf(lrelu(a11[r] * s1 + o1));
        }
    }
}

__global__ __launch_bounds__(256) void head_kernel(
    const unsigned short* __restrict__ xact,
    const unsigned short* __restrict__ Wt0, const unsigned short* __restrict__ Wt1,
    const unsigned short* __restrict__ Wt2, const unsigned short* __restrict__ Wt3,
    const unsigned short* __restrict__ Wt4,
    const float* __restrict__ sc0, const float* __restrict__ sh0,
    const float* __restrict__ sc1, const float* __restrict__ sh1,
    const float* __restrict__ sc2, const float* __restrict__ sh2,
    const float* __restrict__ sc3, const float* __restrict__ sh3,
    const float* __restrict__ sc4, const float* __restrict__ sh4,
    const float* __restrict__ w5r, float* __restrict__ out) {
    extern __shared__ char smem[];
    int tid = threadIdx.x, lane = tid & 31, wave = tid >> 5;
    long row0 = (long)blockIdx.x * MTILE;

    // Stage input rows [64 x 96] (cols 80..95 zero-padded) into buffer 0
    for (int i = tid; i < MTILE * 96; i += 256) {
        int r = i / 96, c = i % 96;
        unsigned short vv = (c < 80) ? xact[(row0 + r) * 80 + c] : (unsigned short)0;
        *(unsigned short*)(smem + (r * STR + c) * 2) = vv;
    }
    __syncthreads();
    mlp_layer<96, 512>(smem, 0, BUFB, Wt0, sc0, sh0, lane, wave);
    __syncthreads();
    mlp_layer<512, 1024>(smem, BUFB, 0, Wt1, sc1, sh1, lane, wave);
    __syncthreads();
    mlp_layer<1024, 1024>(smem, 0, BUFB, Wt2, sc2, sh2, lane, wave);
    __syncthreads();
    mlp_layer<1024, 1024>(smem, BUFB, 0, Wt3, sc3, sh3, lane, wave);
    __syncthreads();
    mlp_layer<1024, 128>(smem, 0, BUFB, Wt4, sc4, sh4, lane, wave);
    __syncthreads();
    // lin5 + mean collapsed to a 128-dot
    if (tid < MTILE) {
        const unsigned short* rp = (const unsigned short*)(smem + BUFB) + tid * STR;
        float acc = w5r[128];
        for (int k = 0; k < 128; ++k) acc += bf2f(rp[k]) * w5r[k];
        out[row0 + tid] = acc;
    }
}

// ---------------------------------------------------------------------------
// Host-side orchestration
// ---------------------------------------------------------------------------
extern "C" void kernel_launch(void* const* d_in, const int* in_sizes, int n_in,
                              void* d_out, int out_size, void* d_ws, size_t ws_size,
                              hipStream_t stream) {
    (void)in_sizes; (void)n_in; (void)out_size; (void)ws_size;
    int p = 0;
    const float* node_feat  = (const float*)d_in[p++];
    const float* edge_feat  = (const float*)d_in[p++];
    const int*   edge_index = (const int*)d_in[p++];

    GArgs ga;
    ga.nf = node_feat; ga.ef = edge_feat; ga.ei = edge_index;
    ga.new_ = (const float*)d_in[p++]; ga.neb = (const float*)d_in[p++];
    ga.eew  = (const float*)d_in[p++]; ga.eeb = (const float*)d_in[p++];
    for (int l = 0; l < LL; ++l)
        for (int s = 0; s < 16; ++s) ga.conv[l][s] = (const float*)d_in[p++];

    // poolin + bind params
    const float *Wl[6], *bl[6], *bg[5], *bb[5], *bm[5], *bv[5];
    // L0 = poolin
    Wl[0] = (const float*)d_in[p++]; bl[0] = (const float*)d_in[p++];
    bg[0] = (const float*)d_in[p++]; bb[0] = (const float*)d_in[p++];
    bm[0] = (const float*)d_in[p++]; bv[0] = (const float*)d_in[p++];
    // L1..L4 = bind lin1..lin4 (+bn), L5 = lin5 (no bn)
    for (int l = 1; l <= 4; ++l) {
        Wl[l] = (const float*)d_in[p++]; bl[l] = (const float*)d_in[p++];
        bg[l] = (const float*)d_in[p++]; bb[l] = (const float*)d_in[p++];
        bm[l] = (const float*)d_in[p++]; bv[l] = (const float*)d_in[p++];
    }
    Wl[5] = (const float*)d_in[p++]; bl[5] = (const float*)d_in[p++];

    // Workspace layout
    char* w = (char*)d_ws;
    size_t off = 0;
    unsigned short* xact = (unsigned short*)(w + off); off += (size_t)BB * 80 * 2;
    const int Ks[5]    = {80, 512, 1024, 1024, 1024};
    const int Kpads[5] = {96, 512, 1024, 1024, 1024};
    const int Ns[5]    = {512, 1024, 1024, 1024, 128};
    unsigned short* Wt[5]; float* sc[5]; float* sh[5];
    for (int l = 0; l < 5; ++l) {
        Wt[l] = (unsigned short*)(w + off); off += (size_t)Kpads[l] * Ns[l] * 2;
    }
    for (int l = 0; l < 5; ++l) {
        sc[l] = (float*)(w + off); off += (size_t)Ns[l] * 4;
        sh[l] = (float*)(w + off); off += (size_t)Ns[l] * 4;
    }
    float* w5r = (float*)(w + off); off += 132 * 4;

    // Weight prep (independent of GNN; stream order serializes before head)
    for (int l = 0; l < 5; ++l) {
        int total = Kpads[l] * Ns[l] + Ns[l];
        prep_layer<<<(total + 255) / 256, 256, 0, stream>>>(
            Wl[l], bl[l], bg[l], bb[l], bm[l], bv[l],
            Wt[l], sc[l], sh[l], Ks[l], Kpads[l], Ns[l]);
    }
    prep_final<<<1, 256, 0, stream>>>(Wl[5], bl[5], w5r);

    // GNN: 32 batch rows per workgroup
    ga.xout = xact;
    gnn_kernel<<<BB / 32, 256, 0, stream>>>(ga);

    // MLP head: 64 batch rows per workgroup, 258 KB LDS ping-pong
    head_kernel<<<BB / MTILE, 256, HEAD_LDS, stream>>>(
        xact, Wt[0], Wt[1], Wt[2], Wt[3], Wt[4],
        sc[0], sh[0], sc[1], sh[1], sc[2], sh[2], sc[3], sh[3], sc[4], sh[4],
        w5r, (float*)d_out);
}
